// ChiralEmbeddingModel_14774687498351
// MI455X (gfx1250) — compile-verified, matrix operand order checked by reading, not run on hardware
//
#include <hip/hip_runtime.h>
#include <hip/hip_bf16.h>
#include <math.h>

// ---------------- types ----------------
typedef __bf16 bf16_t;
typedef bf16_t v16bf __attribute__((ext_vector_type(16)));
typedef float  v8f   __attribute__((ext_vector_type(8)));

union Frag { uint4 q[2]; v16bf f; };

// ---------------- constants ----------------
#define NATOMS 131072
#define INVD   256
#define HD     512
#define KD     64
#define TA     64            // atoms per workgroup
#define EQROWS 192           // 3*TA
#define S_INV  264           // LDS strides (elements); rows 16B-aligned
#define S_EQ   264
#define S_H    520
#define S_X    68            // f32 x-buffer stride
#define S_CHI  68

#define WS_W0  0             // ws (ushort) offsets: col-major bf16 weights
#define WS_W1  16384
#define WS_W2  32768
#define WS_G1  49152         // [n<512][m<256]
#define WS_G2  180224        // [n<64][h<512]
#define WS_TOT 212992

// LDS map (bytes):
//   0      : stats (mean 1K, rstd 1K, gamma 1K, rms 256)        -> 3328
//   3328   : inv_s  bf16 64x264                                  -> 37120
//   37120  : eqA    bf16 192x264 (dead after GEMM-1; chi_s/gate_s alias here)
//   138496 : x0/y1/y2 f32 3x192x68 (dead after chi; hA bf16 64x520 aliases)
//   end    : 295168
#define OFF_INV  3328
#define OFF_EQ   37120
#define OFF_CHI  37120
#define OFF_GATE 54528
#define OFF_X    138496
#define SMEM_BYTES 295168

__device__ __forceinline__ unsigned short f2bf(float x) {
  unsigned int u = __builtin_bit_cast(unsigned int, x);
  unsigned int r = (u + 0x7FFFu + ((u >> 16) & 1u)) >> 16;
  return (unsigned short)r;
}

// WMMA bf16 fragment load (A row-major [row][k]; B stored col-major [n][k] so
// the identical lane/k pattern applies). Per ISA 16-bit layout: lanes 0-15 are
// rows/cols 0-15, k in {kh..kh+7} U {16+kh..16+kh+7}, kh = 8*(lane>>4).
// Consecutive-k pairs are contiguous -> two b128 loads.
__device__ __forceinline__ v16bf load_frag(const unsigned short* base, int row0,
                                           int stride, int kbase, int lane) {
  int rr = row0 + (lane & 15);
  int kh = (lane >> 4) << 3;
  const unsigned short* p = base + rr * stride + kbase + kh;
  Frag fr;
  fr.q[0] = *reinterpret_cast<const uint4*>(p);
  fr.q[1] = *reinterpret_cast<const uint4*>(p + 16);
  return fr.f;
}

#define WMMA_BF16(A, B, C) \
  __builtin_amdgcn_wmma_f32_16x16x32_bf16(false, (A), false, (B), (short)0, (C), false, false)

// ---------------- prologue: fold + transpose + bf16-ify weights ----------------
__global__ void prep_weights(const float* __restrict__ W0, const float* __restrict__ W1,
                             const float* __restrict__ W2, const float* __restrict__ wc,
                             const float* __restrict__ wd, const float* __restrict__ gw1,
                             const float* __restrict__ gw2, unsigned short* __restrict__ ws) {
  const float SL = 0.0625f; // 1/sqrt(256)
  for (int idx = blockIdx.x * blockDim.x + threadIdx.x; idx < WS_TOT;
       idx += gridDim.x * blockDim.x) {
    if (idx < WS_W1) {                       // W0' = s_lin*W0, col-major
      int n = idx >> 8, m = idx & 255;
      ws[idx] = f2bf(W0[m * 64 + n] * SL);
    } else if (idx < WS_G1) {                // W1' = s_lin*W1@wcT ; W2' = s_lin*W2@wdT
      int i = idx - WS_W1;
      const float* Wsrc = (i < 16384) ? W1 : W2;
      const float* wm   = (i < 16384) ? wc : wd;
      i &= 16383;
      int n = i >> 8, m = i & 255;
      float s = 0.f;
      for (int v = 0; v < 64; ++v) s += Wsrc[m * 64 + v] * wm[n * 64 + v];
      ws[idx] = f2bf(s * SL);
    } else if (idx < WS_G2) {                // g_w1 col-major
      int i = idx - WS_G1;
      int n = i >> 8, m = i & 255;
      ws[idx] = f2bf(gw1[m * 512 + n]);
    } else {                                 // g_w2 col-major
      int i = idx - WS_G2;
      int n = i >> 9, h = i & 511;
      ws[idx] = f2bf(gw2[h * 64 + n]);
    }
  }
}

// ---------------- main fused kernel ----------------
__global__ void __launch_bounds__(256, 1)
chiral_main(const float* __restrict__ emb, const float* __restrict__ mean_inv,
            const float* __restrict__ std_inv, const float* __restrict__ rms_gamma,
            const float* __restrict__ ln_w, const float* __restrict__ ln_b,
            const float* __restrict__ g_b1, const float* __restrict__ g_b2,
            const unsigned short* __restrict__ wsb, float* __restrict__ out) {
  extern __shared__ char smem[];
  float* mean_s = (float*)smem;
  float* rstd_s = mean_s + 256;
  float* gam_s  = rstd_s + 256;
  float* rms_s  = gam_s + 256;
  unsigned short* inv_s = (unsigned short*)(smem + OFF_INV);   // 64x264 bf16
  unsigned short* eqA   = (unsigned short*)(smem + OFF_EQ);    // 192x264 bf16
  float* chi_s  = (float*)(smem + OFF_CHI);                    // 64x68 f32 (aliases eqA)
  float* gate_s = (float*)(smem + OFF_GATE);                   // 64x68 f32 (aliases eqA)
  float* xf     = (float*)(smem + OFF_X);                      // 3x192x68 f32
  unsigned short* hA = (unsigned short*)(smem + OFF_X);        // 64x520 bf16 (aliases xf)

  const int tid  = threadIdx.x;
  const int lane = tid & 31;
  const int wid  = tid >> 5;
  const int a0   = blockIdx.x * TA;

  // ---- stage stats (256 threads, one elem each)
  mean_s[tid] = mean_inv[tid];
  rstd_s[tid] = 1.0f / std_inv[tid];
  gam_s[tid]  = rms_gamma[tid];
  __syncthreads();

  // ---- per-atom RMS over the 768 eq values (8 atoms per wave)
  for (int a = wid; a < TA; a += 8) {
    const float* row = emb + (size_t)(a0 + a) * 1024 + INVD;
    float s = 0.f;
    for (int i = lane; i < 768; i += 32) { float v = row[i]; s += v * v; }
    for (int off = 16; off > 0; off >>= 1) s += __shfl_xor(s, off, 32);
    if (lane == 0) rms_s[a] = rsqrtf(s * (1.0f / 256.0f) + 1e-6f);
  }
  __syncthreads();

  // ---- stage normalized inv (bf16) and eq as A-matrix [atom*3+j][m]
  for (int e = tid; e < TA * INVD; e += 256) {
    int a = e >> 8, c = e & 255;
    float v = (emb[(size_t)(a0 + a) * 1024 + c] - mean_s[c]) * rstd_s[c];
    inv_s[a * S_INV + c] = f2bf(v);
  }
  for (int e = tid; e < TA * 768; e += 256) {
    int a = e / 768, r = e - a * 768;
    int m = r / 3, j = r - m * 3;
    float v = emb[(size_t)(a0 + a) * 1024 + INVD + r] * gam_s[m] * rms_s[a];
    eqA[(a * 3 + j) * S_EQ + m] = f2bf(v);
  }
  __syncthreads();

  // ---- GEMM-1: x0/y1/y2 (192x64 f32) = eqA(192x256) @ W'{0,1,2}(256x64)
  // One A-fragment feeds three WMMAs (shared across the 3 weight matrices).
  {
    float* X0 = xf;
    float* Y1 = xf + EQROWS * S_X;
    float* Y2 = xf + 2 * EQROWS * S_X;
    for (int t = wid; t < 48; t += 8) {   // 12 row-tiles x 4 col-tiles
      int rt = t >> 2, ct = t & 3;
      v8f a0v = {}, a1v = {}, a2v = {};
      for (int kk = 0; kk < 8; ++kk) {
        v16bf af = load_frag(eqA, rt * 16, S_EQ, kk * 32, lane);
        v16bf b0 = load_frag(wsb + WS_W0, ct * 16, 256, kk * 32, lane);
        v16bf b1 = load_frag(wsb + WS_W1, ct * 16, 256, kk * 32, lane);
        v16bf b2 = load_frag(wsb + WS_W2, ct * 16, 256, kk * 32, lane);
        a0v = WMMA_BF16(af, b0, a0v);
        a1v = WMMA_BF16(af, b1, a1v);
        a2v = WMMA_BF16(af, b2, a2v);
      }
      int n  = ct * 16 + (lane & 15);
      int mb = rt * 16 + ((lane >> 4) << 3);
#pragma unroll
      for (int r = 0; r < 8; ++r) {
        X0[(mb + r) * S_X + n] = a0v[r];
        Y1[(mb + r) * S_X + n] = a1v[r];
        Y2[(mb + r) * S_X + n] = a2v[r];
      }
    }
  }
  __syncthreads();

  // ---- chi = dot(cross(x0,y1), y2) / (sqrt(2K)*sqrt(3K))   (writes alias eqA)
  for (int p = tid; p < TA * KD; p += 256) {
    int a = p >> 6, u = p & 63;
    const float* X0 = xf;
    const float* Y1 = xf + EQROWS * S_X;
    const float* Y2 = xf + 2 * EQROWS * S_X;
    int rb = a * 3;
    float x0[3], y1[3], y2[3];
#pragma unroll
    for (int j = 0; j < 3; ++j) {
      x0[j] = X0[(rb + j) * S_X + u];
      y1[j] = Y1[(rb + j) * S_X + u];
      y2[j] = Y2[(rb + j) * S_X + u];
    }
    float c0 = x0[1] * y1[2] - x0[2] * y1[1];
    float c1 = x0[2] * y1[0] - x0[0] * y1[2];
    float c2 = x0[0] * y1[1] - x0[1] * y1[0];
    chi_s[a * S_CHI + u] = (c0 * y2[0] + c1 * y2[1] + c2 * y2[2]) * 6.3788743e-3f;
  }
  __syncthreads();

  // ---- GEMM-2: hA(64x512 bf16) = silu(inv_s @ g_w1 + b1)   (hA aliases xf)
  const unsigned short* G1 = wsb + WS_G1;
  for (int t = wid; t < 128; t += 8) {
    int rt = t >> 5, ct = t & 31;
    v8f acc = {};
    for (int kk = 0; kk < 8; ++kk) {
      v16bf af = load_frag(inv_s, rt * 16, S_INV, kk * 32, lane);
      v16bf bf = load_frag(G1, ct * 16, 256, kk * 32, lane);
      acc = WMMA_BF16(af, bf, acc);
    }
    int n  = ct * 16 + (lane & 15);
    int mb = rt * 16 + ((lane >> 4) << 3);
    float b = g_b1[n];
#pragma unroll
    for (int r = 0; r < 8; ++r) {
      float x = acc[r] + b;
      hA[(mb + r) * S_H + n] = f2bf(x / (1.0f + __expf(-x)));  // SiLU
    }
  }
  __syncthreads();

  // ---- GEMM-3: gate(64x64) = sigmoid(hA @ g_w2 + b2)
  const unsigned short* G2 = wsb + WS_G2;
  for (int t = wid; t < 16; t += 8) {
    int rt = t >> 2, ct = t & 3;
    v8f acc = {};
    for (int kk = 0; kk < 16; ++kk) {
      v16bf af = load_frag(hA, rt * 16, S_H, kk * 32, lane);
      v16bf bf = load_frag(G2, ct * 16, 512, kk * 32, lane);
      acc = WMMA_BF16(af, bf, acc);
    }
    int n  = ct * 16 + (lane & 15);
    int mb = rt * 16 + ((lane >> 4) << 3);
    float b = g_b2[n];
#pragma unroll
    for (int r = 0; r < 8; ++r) {
      float x = acc[r] + b;
      gate_s[(mb + r) * S_CHI + n] = 1.0f / (1.0f + __expf(-x));
    }
  }
  __syncthreads();

  // ---- LayerNorm over K=64 (wave32: 2 elems/lane) + gate multiply + store
  for (int a = wid; a < TA; a += 8) {
    float c0 = chi_s[a * S_CHI + lane];
    float c1 = chi_s[a * S_CHI + lane + 32];
    float s = c0 + c1;
    for (int off = 16; off > 0; off >>= 1) s += __shfl_xor(s, off, 32);
    float mu = s * (1.0f / 64.0f);
    float d0 = c0 - mu, d1 = c1 - mu;
    float v = d0 * d0 + d1 * d1;
    for (int off = 16; off > 0; off >>= 1) v += __shfl_xor(v, off, 32);
    float rstd = rsqrtf(v * (1.0f / 64.0f) + 1e-5f);
    float* orow = out + (size_t)(a0 + a) * 64;
    int u0 = lane, u1 = lane + 32;
    orow[u0] = gate_s[a * S_CHI + u0] * (d0 * rstd * ln_w[u0] + ln_b[u0]);
    orow[u1] = gate_s[a * S_CHI + u1] * (d1 * rstd * ln_w[u1] + ln_b[u1]);
  }
}

// ---------------- launch ----------------
extern "C" void kernel_launch(void* const* d_in, const int* in_sizes, int n_in,
                              void* d_out, int out_size, void* d_ws, size_t ws_size,
                              hipStream_t stream) {
  const float* emb   = (const float*)d_in[0];
  const float* meanv = (const float*)d_in[1];
  const float* stdv  = (const float*)d_in[2];
  const float* gam   = (const float*)d_in[3];
  const float* W0    = (const float*)d_in[4];
  const float* W1    = (const float*)d_in[5];
  const float* W2    = (const float*)d_in[6];
  const float* wc    = (const float*)d_in[7];
  const float* wd    = (const float*)d_in[8];
  const float* lnw   = (const float*)d_in[9];
  const float* lnb   = (const float*)d_in[10];
  const float* gw1   = (const float*)d_in[11];
  const float* gb1   = (const float*)d_in[12];
  const float* gw2   = (const float*)d_in[13];
  const float* gb2   = (const float*)d_in[14];
  unsigned short* ws = (unsigned short*)d_ws;
  float* outp        = (float*)d_out;

  (void)in_sizes; (void)n_in; (void)out_size; (void)ws_size;
  (void)hipFuncSetAttribute((const void*)chiral_main,
                            hipFuncAttributeMaxDynamicSharedMemorySize, SMEM_BYTES);

  prep_weights<<<832, 256, 0, stream>>>(W0, W1, W2, wc, wd, gw1, gw2, ws);
  chiral_main<<<NATOMS / TA, 256, SMEM_BYTES, stream>>>(
      emb, meanv, stdv, gam, lnw, lnb, gb1, gb2, ws, outp);
}
